// VectorQuantizer_69458211110925
// MI455X (gfx1250) — matrix operand order repaired and checked by reading, not compiled, not run on hardware
//
#include <hip/hip_runtime.h>
#include <hip/hip_bf16.h>

// VQ-VAE vector quantization for MI455X (gfx1250, wave32, WMMA).
//
// Roofline: 2*N*K*C = 137 GFLOP matmul vs ~75MB HBM -> compute bound.
// Strategy: bf16 hi/lo split (3x v_wmma_f32_16x16x32_bf16 per tile) gives
// near-fp32 argmin accuracy at ~2.7x the fp32-WMMA rate. Weight (8MB as
// bf16 hi+lo) is L2-resident (192MB L2), so per-wave restreaming is cheap.
// A-fragments for each wave's 16 rows live in VGPRs for the entire K sweep.

typedef __attribute__((ext_vector_type(16))) __bf16 v16bf;
typedef __attribute__((ext_vector_type(8)))  float  v8f;

#define VQ_B    4
#define VQ_C    256
#define VQ_K    8192
#define VQ_THW  8192            // T*H*W = 8*32*32
#define VQ_N    32768           // B*THW
#define VQ_NUMEL 8388608        // B*C*T*H*W

// ---------------------------------------------------------------------------
// Kernel 1: convert codebook to bf16 hi/lo, compute ||w_k||^2.
// One block per codebook row, one thread per channel.
// ---------------------------------------------------------------------------
__global__ void __launch_bounds__(256)
vq_prep_weight(const float* __restrict__ w,
               __bf16* __restrict__ whi, __bf16* __restrict__ wlo,
               float* __restrict__ wnorm)
{
    const int k = blockIdx.x;
    const int c = threadIdx.x;
    float v = w[(size_t)k * VQ_C + c];
    __bf16 h = (__bf16)v;
    __bf16 l = (__bf16)(v - (float)h);
    whi[(size_t)k * VQ_C + c] = h;
    wlo[(size_t)k * VQ_C + c] = l;

    __shared__ float red[256];
    red[c] = v * v;
    __syncthreads();
    for (int s = 128; s > 0; s >>= 1) {
        if (c < s) red[c] += red[c + s];
        __syncthreads();
    }
    if (c == 0) wnorm[k] = red[0];
}

// ---------------------------------------------------------------------------
// Kernel 2: fused GEMM + argmin. Each wave owns 16 rows of z_flat and sweeps
// all K=8192 codebook entries in 16-wide tiles; 8 c-chunks x 3 hi/lo WMMAs
// per tile. argmin over columns is carried per-lane, then reduced with
// shfl_xor within each 16-lane half (C-matrix layout: lanes 0-15 = rows 0-7,
// lanes 16-31 = rows 8-15).
// ---------------------------------------------------------------------------
__global__ void __launch_bounds__(256)
vq_argmin(const float* __restrict__ z,
          const __bf16* __restrict__ whi, const __bf16* __restrict__ wlo,
          const float* __restrict__ wnorm, int* __restrict__ idx_out)
{
    const int lane   = threadIdx.x & 31;
    const int wgid   = blockIdx.x * 8 + (threadIdx.x >> 5); // global wave id
    const int r0     = wgid * 16;                           // first row n
    const int b      = r0 / VQ_THW;
    const int thw0   = r0 % VQ_THW;                         // tile stays in one b (8192%16==0)
    const int m      = lane & 15;                           // A-matrix row M
    const bool hiH   = lane >= 16;

    // --- Load A fragments (16x32 bf16 per chunk), hi/lo split, into VGPRs.
    // A layout (matches B): lanes 0-15: elements 0..15 = K 0..15;
    //                       lanes 16-31: elements 0..15 = K 16..31.
    const int akb = hiH ? 16 : 0;
    v16bf ahi[8], alo[8];
    const float* zrow = z + (size_t)b * VQ_C * VQ_THW + thw0 + m; // stride THW per c
#pragma unroll
    for (int cc = 0; cc < 8; ++cc) {
#pragma unroll
        for (int j = 0; j < 8; ++j) {
            float v0 = zrow[(size_t)(cc * 32 + akb + j) * VQ_THW];
            __bf16 h0 = (__bf16)v0;
            ahi[cc][j]     = h0;
            alo[cc][j]     = (__bf16)(v0 - (float)h0);
            float v1 = zrow[(size_t)(cc * 32 + 8 + akb + j) * VQ_THW];
            __bf16 h1 = (__bf16)v1;
            ahi[cc][8 + j] = h1;
            alo[cc][8 + j] = (__bf16)(v1 - (float)h1);
        }
    }

    // B-matrix layout (32x16 bf16): lane holds column n = lane%16,
    // K = (lane<16 ? 0..15 : 16..31) -> 16 *consecutive* bf16 per lane.
    const int n16 = lane & 15;
    const int bkb = hiH ? 16 : 0;

    float bestv[8];
    int   besti[8];
#pragma unroll
    for (int r = 0; r < 8; ++r) { bestv[r] = 3.4e38f; besti[r] = 0; }

    for (int kt = 0; kt < VQ_K / 16; ++kt) {
        const int krow = kt * 16 + n16;
        const __bf16* bh = whi + (size_t)krow * VQ_C + bkb;
        const __bf16* bl = wlo + (size_t)krow * VQ_C + bkb;
        v8f acc = {};
#pragma unroll
        for (int cc = 0; cc < 8; ++cc) {
            v16bf bhv = *(const v16bf*)(bh + cc * 32);   // 32B aligned
            v16bf blv = *(const v16bf*)(bl + cc * 32);
            acc = __builtin_amdgcn_wmma_f32_16x16x32_bf16(
                false, ahi[cc], false, bhv, (short)0, acc, false, false);
            acc = __builtin_amdgcn_wmma_f32_16x16x32_bf16(
                false, ahi[cc], false, blv, (short)0, acc, false, false);
            acc = __builtin_amdgcn_wmma_f32_16x16x32_bf16(
                false, alo[cc], false, bhv, (short)0, acc, false, false);
        }
        const float wn = wnorm[krow];
        // dist(n,k) = ||z||^2 + ||w||^2 - 2 z.w ; drop row-constant ||z||^2.
#pragma unroll
        for (int r = 0; r < 8; ++r) {
            float val = __builtin_fmaf(-2.0f, acc[r], wn);
            if (val < bestv[r]) { bestv[r] = val; besti[r] = krow; }
        }
    }

    // Cross-lane (min,argmin) within each 16-lane half; first-index tiebreak.
#pragma unroll
    for (int r = 0; r < 8; ++r) {
        float v = bestv[r]; int i = besti[r];
#pragma unroll
        for (int off = 1; off < 16; off <<= 1) {
            float ov = __shfl_xor(v, off, 32);
            int   oi = __shfl_xor(i, off, 32);
            if (ov < v || (ov == v && oi < i)) { v = ov; i = oi; }
        }
        bestv[r] = v; besti[r] = i;
    }
    if ((lane & 15) == 0) {
        const int rowbase = r0 + (hiH ? 8 : 0);
#pragma unroll
        for (int r = 0; r < 8; ++r) idx_out[rowbase + r] = besti[r];
    }
}

// ---------------------------------------------------------------------------
// Kernel 3: gather z_q = weight[idx], write in [B,C,T,H,W] layout, and
// produce per-block partial sums of (z_q - z)^2 (fixed-order, deterministic).
// blockIdx.y = channel c, blockIdx.x = chunk of 256 consecutive n
// -> coalesced z reads / z_q writes along thw.
// ---------------------------------------------------------------------------
__global__ void __launch_bounds__(256)
vq_gather(const float* __restrict__ z, const float* __restrict__ w,
          const int* __restrict__ idx, float* __restrict__ zq,
          float* __restrict__ partials)
{
    const int c   = blockIdx.y;
    const int n   = blockIdx.x * 256 + threadIdx.x;
    const int b   = n / VQ_THW;
    const int thw = n % VQ_THW;
    const int k   = idx[n];
    const float wv = w[(size_t)k * VQ_C + c];
    const size_t zo = ((size_t)b * VQ_C + c) * VQ_THW + thw;
    const float zv = z[zo];
    zq[zo] = wv;                       // z_q_st == z + sg(z_q - z) == z_q
    const float d = wv - zv;

    __shared__ float red[256];
    red[threadIdx.x] = d * d;
    __syncthreads();
    for (int s = 128; s > 0; s >>= 1) {
        if (threadIdx.x < s) red[threadIdx.x] += red[threadIdx.x + s];
        __syncthreads();
    }
    if (threadIdx.x == 0)
        partials[(size_t)c * (VQ_N / 256) + blockIdx.x] = red[0];
}

// ---------------------------------------------------------------------------
// Kernel 4: deterministic final reduction.
// vq_loss = (1 + 0.25) * mean((z_q - z)^2)  (both loss terms equal in value).
// ---------------------------------------------------------------------------
__global__ void __launch_bounds__(256)
vq_loss_reduce(const float* __restrict__ partials, float* __restrict__ loss_out)
{
    __shared__ float red[256];
    float s = 0.f;
    for (int i = threadIdx.x; i < 32768; i += 256) s += partials[i];
    red[threadIdx.x] = s;
    __syncthreads();
    for (int t = 128; t > 0; t >>= 1) {
        if (threadIdx.x < t) red[threadIdx.x] += red[threadIdx.x + t];
        __syncthreads();
    }
    if (threadIdx.x == 0)
        loss_out[0] = 1.25f * red[0] / (float)VQ_NUMEL;
}

// ---------------------------------------------------------------------------
extern "C" void kernel_launch(void* const* d_in, const int* in_sizes, int n_in,
                              void* d_out, int out_size, void* d_ws, size_t ws_size,
                              hipStream_t stream)
{
    const float* z = (const float*)d_in[0];   // [B,C,T,H,W] fp32
    const float* w = (const float*)d_in[1];   // [K,C] fp32

    // d_out: z_q_st (8388608 f32) | vq_loss (1 f32) | indices (32768 i32)
    float* zq   = (float*)d_out;
    float* loss = (float*)d_out + VQ_NUMEL;
    int*   idx  = (int*)d_out + VQ_NUMEL + 1;

    // Workspace layout (~8.2 MB):
    char* ws = (char*)d_ws;
    __bf16* whi   = (__bf16*)ws;                                   // 4 MB
    __bf16* wlo   = (__bf16*)(ws + (size_t)VQ_K * VQ_C * 2);       // 4 MB
    float*  wnorm = (float*)(ws + (size_t)VQ_K * VQ_C * 4);        // 32 KB
    float*  parts = (float*)(ws + (size_t)VQ_K * VQ_C * 4 + VQ_K * 4); // 128 KB

    vq_prep_weight<<<VQ_K, 256, 0, stream>>>(w, whi, wlo, wnorm);
    // 2048 waves x 16 rows = 32768 rows; 8 waves per 256-thread block.
    vq_argmin<<<VQ_N / 16 / 8, 256, 0, stream>>>(z, whi, wlo, wnorm, idx);
    dim3 g(VQ_N / 256, VQ_C);
    vq_gather<<<g, 256, 0, stream>>>(z, w, idx, zq, parts);
    vq_loss_reduce<<<1, 256, 0, stream>>>(parts, loss);
}
